// DeformConvBlock_1254130450589
// MI455X (gfx1250) — matrix operand order repaired and checked by reading, not compiled, not run on hardware
//
#include <hip/hip_runtime.h>

typedef __attribute__((ext_vector_type(16))) __bf16 v16bf;
typedef __attribute__((ext_vector_type(8)))  float  v8f;
typedef int v2i __attribute__((vector_size(8)));

union Frag16 { v16bf v; unsigned u[8]; };

#if defined(__gfx1250__)
#if __has_builtin(__builtin_amdgcn_global_load_async_to_lds_b64)
#define USE_ASYNC_LDS 1
#endif
#endif

#ifdef USE_ASYNC_LDS
static __device__ __forceinline__ void async_b64(const __bf16* g, __bf16* l) {
  __builtin_amdgcn_global_load_async_to_lds_b64(
      (__attribute__((address_space(1))) v2i*)g,
      (__attribute__((address_space(3))) v2i*)l, 0, 0);
}
static __device__ __forceinline__ void wait_async() {
#if __has_builtin(__builtin_amdgcn_s_wait_asynccnt)
  __builtin_amdgcn_s_wait_asynccnt(0);
#else
  asm volatile("s_wait_asynccnt 0" ::: "memory");
#endif
}
#endif

static __device__ __forceinline__ unsigned pack_bf16(float lo, float hi) {
  unsigned short a = __builtin_bit_cast(unsigned short, (__bf16)lo);
  unsigned short b = __builtin_bit_cast(unsigned short, (__bf16)hi);
  return (unsigned)a | ((unsigned)b << 16);
}

// 16-bit A 16x32 — lanes 0-15 hold K 0..15, lanes 16-31 hold K 16..31 (matches B)
static __device__ __forceinline__ int kmapA(int p, int laneHi) {
  return 2 * p + 16 * laneHi;
}
// 16-bit B 32x16 — lanes 0-15 hold K 0..15, lanes 16-31 hold K 16..31
static __device__ __forceinline__ int kmapB(int p, int laneHi) {
  return 2 * p + 16 * laneHi;
}

// ---------------- Kernel A: NCHW -> NHWC (f32 + bf16 copies) -------------
__global__ __launch_bounds__(256) void k_transpose(const float* __restrict__ x,
                                                   float* __restrict__ xf,
                                                   __bf16* __restrict__ xb) {
  __shared__ float tile[64][65];
  int cc = blockIdx.x & 3;          // channel chunk of 64
  int bh = blockIdx.x >> 2;         // b*64 + h
  int b = bh >> 6, h = bh & 63;
  int c0 = cc * 64;
  for (int i = threadIdx.x; i < 4096; i += 256) {
    int c = i >> 6, w = i & 63;     // coalesced along w
    tile[c][w] = x[(((size_t)b * 256 + c0 + c) * 64 + h) * 64 + w];
  }
  __syncthreads();
  size_t obase = (((size_t)b * 64 + h) * 64) * 256;
  for (int i = threadIdx.x; i < 4096; i += 256) {
    int w = i >> 6, c = i & 63;     // coalesced along c
    float v = tile[c][w];
    size_t o = obase + (size_t)w * 256 + c0 + c;
    xf[o] = v;
    xb[o] = (__bf16)v;
  }
}

// ---------------- Kernel W1: pack offset-conv weights --------------------
// Boff[n][K] (N-major), N padded 72->80, K = tap*256 + c
__global__ __launch_bounds__(256) void k_pack_off(const float* __restrict__ w_off,
                                                  __bf16* __restrict__ Boff) {
  int i = blockIdx.x * 256 + threadIdx.x;
  if (i >= 80 * 2304) return;
  int n = i / 2304, K = i % 2304;
  int tap = K >> 8, c = K & 255;
  float v = (n < 72) ? w_off[((size_t)n * 256 + c) * 9 + tap] : 0.f;
  Boff[i] = (__bf16)v;
}

// ---------------- Kernel W2: pack deform-conv weights --------------------
// Bdef[g][tap][n][cg]
__global__ __launch_bounds__(256) void k_pack_def(const float* __restrict__ w_def,
                                                  __bf16* __restrict__ Bdef) {
  int i = blockIdx.x * 256 + threadIdx.x;
  if (i >= 4 * 9 * 64 * 64) return;
  int cg = i & 63;
  int n  = (i >> 6) & 63;
  int gk = i >> 12;
  int k  = gk % 9;
  int g  = gk / 9;
  Bdef[i] = (__bf16)w_def[((size_t)(g * 64 + n) * 64 + cg) * 9 + k];
}

// ---------------- Kernel 1: offset conv (implicit GEMM, bf16 WMMA) -------
// block = one image row: M=64 pixels, N=80 (72 valid), K=2304, K-chunk=64
__global__ __launch_bounds__(128) void k_conv_off(const __bf16* __restrict__ xb,
                                                  const __bf16* __restrict__ Boff,
                                                  const float* __restrict__ b_off,
                                                  float* __restrict__ offs) {
  __shared__ __bf16 ldsA[64 * 68];   // [pixel][K-chunk 64], stride 68
  __shared__ __bf16 ldsB[80 * 68];   // [n][K-chunk 64], stride 68
  int b = blockIdx.x >> 6, h = blockIdx.x & 63;
  int wid = threadIdx.x >> 5, lane = threadIdx.x & 31;
  int lan15 = lane & 15, laneHi = lane >> 4;
  int pix = threadIdx.x >> 1, half = threadIdx.x & 1;

  v8f acc[5];
  for (int i = 0; i < 5; i++)
    for (int r = 0; r < 8; r++) acc[i][r] = 0.f;

  for (int kb = 0; kb < 2304; kb += 64) {
    // ---- issue B tile (80 x 64 bf16) first: async DMA to LDS if available
    for (int i = threadIdx.x; i < 1280; i += 128) {  // 80 rows x 16 qwords
      int n = i >> 4, j = i & 15;
      const __bf16* src = Boff + (size_t)n * 2304 + kb + 4 * j;
      __bf16* dst = ldsB + n * 68 + 4 * j;
#ifdef USE_ASYNC_LDS
      async_b64(src, dst);
#else
      *(uint2*)dst = *(const uint2*)src;
#endif
    }
    if (threadIdx.x < 80 && kb + 64 < 2304)
      __builtin_prefetch(Boff + (size_t)threadIdx.x * 2304 + kb + 64, 0, 0);

    // ---- A tile: zero-padded 3x3 patch slice (64 pixels x 64 channels)
    int tap = kb >> 8, c0 = kb & 255;
    int dy = tap / 3 - 1, dx = tap % 3 - 1;
    int ys = h + dy, xs = pix + dx;
    bool ok = ((unsigned)ys < 64u) && ((unsigned)xs < 64u);
    {
      uint2* dst = (uint2*)(ldsA + pix * 68 + half * 32);
      if (ok) {
        const uint2* src = (const uint2*)(xb + ((((size_t)b * 64 + ys) * 64 + xs) * 256 + c0 + half * 32));
        #pragma unroll
        for (int j = 0; j < 8; j++) dst[j] = src[j];
      } else {
        uint2 z; z.x = 0u; z.y = 0u;
        #pragma unroll
        for (int j = 0; j < 8; j++) dst[j] = z;
      }
    }
#ifdef USE_ASYNC_LDS
    wait_async();
#endif
    __syncthreads();

    int arow = wid * 16 + lan15;
    #pragma unroll
    for (int kc = 0; kc < 2; kc++) {
      Frag16 a;
      #pragma unroll
      for (int p = 0; p < 8; p++)
        a.u[p] = *(const unsigned*)(ldsA + arow * 68 + kc * 32 + kmapA(p, laneHi));
      #pragma unroll
      for (int nt = 0; nt < 5; nt++) {
        Frag16 bm;
        int n = nt * 16 + lan15;
        #pragma unroll
        for (int p = 0; p < 8; p++)
          bm.u[p] = *(const unsigned*)(ldsB + n * 68 + kc * 32 + kmapB(p, laneHi));
        acc[nt] = __builtin_amdgcn_wmma_f32_16x16x32_bf16(
            false, a.v, false, bm.v, (short)0, acc[nt], false, false);
      }
    }
    __syncthreads();
  }

  // offsets stored [B][H][W][72] (f32)
  size_t obase = (((size_t)b * 64 + h) * 64) * 72;
  #pragma unroll
  for (int nt = 0; nt < 5; nt++) {
    int n = nt * 16 + lan15;
    if (n < 72) {
      float bias = b_off[n];
      #pragma unroll
      for (int r = 0; r < 8; r++) {
        int w = wid * 16 + r + 8 * laneHi;
        offs[obase + (size_t)w * 72 + n] = acc[nt][r] + bias;
      }
    }
  }
}

static __device__ __forceinline__ void corner_acc(const float* __restrict__ xbase,
                                                  int y, int x, float w, float* v) {
  if ((unsigned)y < 64u && (unsigned)x < 64u) {
    const float4* p4 = (const float4*)(xbase + ((size_t)y * 64 + x) * 256);
    #pragma unroll
    for (int i = 0; i < 8; i++) {
      float4 f = p4[i];
      v[4 * i + 0] += w * f.x;
      v[4 * i + 1] += w * f.y;
      v[4 * i + 2] += w * f.z;
      v[4 * i + 3] += w * f.w;
    }
  }
}

// ---------------- Kernel 2: deformable conv (sample + bf16 WMMA) ---------
// block = (b, g, h): M=64 pixels, N=64 outputs, K=576 (9 taps x 64 ch)
__global__ __launch_bounds__(128) void k_deform(const float* __restrict__ xf,
                                                const float* __restrict__ offs,
                                                const __bf16* __restrict__ Bdef,
                                                const float* __restrict__ b_def,
                                                float* __restrict__ out) {
  __shared__ __bf16 ldsA[64 * 72];   // [pixel][cg 64], stride 72
  __shared__ __bf16 ldsB[64 * 72];   // [n][cg 64], stride 72
  __shared__ float  ldsO[64 * 18];   // per-pixel 9x(dy,dx)
  int h = blockIdx.x & 63;
  int g = (blockIdx.x >> 6) & 3;
  int b = blockIdx.x >> 8;
  int wid = threadIdx.x >> 5, lane = threadIdx.x & 31;
  int lan15 = lane & 15, laneHi = lane >> 4;
  int pix = threadIdx.x >> 1, half = threadIdx.x & 1;

  {
    size_t base = (((size_t)b * 64 + h) * 64) * 72 + g * 18;
    for (int i = threadIdx.x; i < 64 * 18; i += 128) {
      int p = i / 18, j = i % 18;
      ldsO[i] = offs[base + (size_t)p * 72 + j];
    }
  }
  v8f acc[4];
  for (int i = 0; i < 4; i++)
    for (int r = 0; r < 8; r++) acc[i][r] = 0.f;
  __syncthreads();

  const float* xbase = xf + (size_t)b * 64 * 64 * 256 + g * 64 + half * 32;

  for (int k = 0; k < 9; k++) {
    int dy = k / 3 - 1, dx = k % 3 - 1;
    {  // issue weight tile [n][cg] 64x64 bf16 first (async DMA if available)
      const __bf16* wsrc = Bdef + (size_t)((g * 9 + k) * 64) * 64;
      for (int i = threadIdx.x; i < 1024; i += 128) {  // 64 rows x 16 qwords
        int n = i >> 4, j = i & 15;
        const __bf16* src = wsrc + (size_t)n * 64 + 4 * j;
        __bf16* dst = ldsB + n * 72 + 4 * j;
#ifdef USE_ASYNC_LDS
        async_b64(src, dst);
#else
        *(uint2*)dst = *(const uint2*)src;
#endif
      }
    }
    {  // bilinear-sample A tile: this thread = pixel `pix`, channels half*32..+31
      float py = ldsO[pix * 18 + 2 * k + 0] + (float)(h + dy);
      float px = ldsO[pix * 18 + 2 * k + 1] + (float)(pix + dx);
      float y0f = floorf(py), x0f = floorf(px);
      int y0 = (int)y0f, x0 = (int)x0f;
      float wy1 = py - y0f, wy0 = 1.f - wy1;
      float wx1 = px - x0f, wx0 = 1.f - wx1;
      float v[32];
      #pragma unroll
      for (int i = 0; i < 32; i++) v[i] = 0.f;
      corner_acc(xbase, y0,     x0,     wy0 * wx0, v);
      corner_acc(xbase, y0,     x0 + 1, wy0 * wx1, v);
      corner_acc(xbase, y0 + 1, x0,     wy1 * wx0, v);
      corner_acc(xbase, y0 + 1, x0 + 1, wy1 * wx1, v);
      #pragma unroll
      for (int i = 0; i < 16; i++)
        *(unsigned*)(ldsA + pix * 72 + half * 32 + 2 * i) = pack_bf16(v[2 * i], v[2 * i + 1]);
    }
#ifdef USE_ASYNC_LDS
    wait_async();
#endif
    __syncthreads();

    int arow = wid * 16 + lan15;
    #pragma unroll
    for (int kc = 0; kc < 2; kc++) {
      Frag16 a;
      #pragma unroll
      for (int p = 0; p < 8; p++)
        a.u[p] = *(const unsigned*)(ldsA + arow * 72 + kc * 32 + kmapA(p, laneHi));
      #pragma unroll
      for (int nt = 0; nt < 4; nt++) {
        Frag16 bm;
        int n = nt * 16 + lan15;
        #pragma unroll
        for (int p = 0; p < 8; p++)
          bm.u[p] = *(const unsigned*)(ldsB + n * 72 + kc * 32 + kmapB(p, laneHi));
        acc[nt] = __builtin_amdgcn_wmma_f32_16x16x32_bf16(
            false, a.v, false, bm.v, (short)0, acc[nt], false, false);
      }
    }
    __syncthreads();
  }

  #pragma unroll
  for (int nt = 0; nt < 4; nt++) {
    int ch = g * 64 + nt * 16 + lan15;
    float bias = b_def[ch];
    size_t cbase = (((size_t)b * 256 + ch) * 64 + h) * 64;
    #pragma unroll
    for (int r = 0; r < 8; r++) {
      int w = wid * 16 + r + 8 * laneHi;
      out[cbase + w] = acc[nt][r] + bias;
    }
  }
}

extern "C" void kernel_launch(void* const* d_in, const int* in_sizes, int n_in,
                              void* d_out, int out_size, void* d_ws, size_t ws_size,
                              hipStream_t stream) {
  const float* x     = (const float*)d_in[0];
  const float* w_off = (const float*)d_in[1];
  const float* b_off = (const float*)d_in[2];
  const float* w_def = (const float*)d_in[3];
  const float* b_def = (const float*)d_in[4];
  float* out = (float*)d_out;

  char* ws = (char*)d_ws;
  float*  xf   = (float*)ws;                      // 33,554,432 B  NHWC f32
  __bf16* xb   = (__bf16*)(ws + 33554432);        // 16,777,216 B  NHWC bf16
  __bf16* Boff = (__bf16*)(ws + 50331648);        //    368,640 B  packed offset weights
  __bf16* Bdef = (__bf16*)(ws + 50700288);        //    294,912 B  packed deform weights
  float*  offs = (float*)(ws + 50995200);         //  9,437,184 B  offsets [B][H][W][72]

  hipLaunchKernelGGL(k_transpose, dim3(2048), dim3(256), 0, stream, x, xf, xb);
  hipLaunchKernelGGL(k_pack_off, dim3(720), dim3(256), 0, stream, w_off, Boff);
  hipLaunchKernelGGL(k_pack_def, dim3(576), dim3(256), 0, stream, w_def, Bdef);
  hipLaunchKernelGGL(k_conv_off, dim3(512), dim3(128), 0, stream, xb, Boff, b_off, offs);
  hipLaunchKernelGGL(k_deform, dim3(2048), dim3(128), 0, stream, xf, offs, Bdef, b_def, out);
}
